// TrajectoryTransformer_12670153523821
// MI455X (gfx1250) — matrix operand order repaired and checked by reading, not compiled, not run on hardware
//
#include <hip/hip_runtime.h>
#include <hip/hip_bf16.h>
#include <math.h>

// Model dims (fixed by the reference)
#define BB 256
#define TT 100
#define IN 6
#define DD 1024
#define HH 8
#define LL 4
#define OUT 256
#define HD 128              // D / H
#define M_TOK (BB * TT)     // 25600 token rows
#define SCALE 0.08838834764831845f  // 1/sqrt(128)

typedef __attribute__((ext_vector_type(16))) _Float16 v16h;
typedef __attribute__((ext_vector_type(8)))  _Float16 h8;
typedef __attribute__((ext_vector_type(4)))  _Float16 h4;
typedef __attribute__((ext_vector_type(8)))  float    v8f;

__device__ __forceinline__ h4 cvt_h4(float4 f) {
  h4 r; r[0] = (_Float16)f.x; r[1] = (_Float16)f.y;
        r[2] = (_Float16)f.z; r[3] = (_Float16)f.w;
  return r;
}
__device__ __forceinline__ v16h cat16(h8 lo, h8 hi) {
  return __builtin_shufflevector(lo, hi, 0,1,2,3,4,5,6,7,8,9,10,11,12,13,14,15);
}
__device__ __forceinline__ v16h pack16(float4 f0, float4 f1, float4 f2, float4 f3) {
  v16h a;
  a[0]=(_Float16)f0.x;  a[1]=(_Float16)f0.y;  a[2]=(_Float16)f0.z;  a[3]=(_Float16)f0.w;
  a[4]=(_Float16)f1.x;  a[5]=(_Float16)f1.y;  a[6]=(_Float16)f1.z;  a[7]=(_Float16)f1.w;
  a[8]=(_Float16)f2.x;  a[9]=(_Float16)f2.y;  a[10]=(_Float16)f2.z; a[11]=(_Float16)f2.w;
  a[12]=(_Float16)f3.x; a[13]=(_Float16)f3.y; a[14]=(_Float16)f3.z; a[15]=(_Float16)f3.w;
  return a;
}

// -------------------------------------------------------------------------
// Embedding: h[b,t,:] = x[b,t,:6] @ Wi + bi + PE(t, :)
// -------------------------------------------------------------------------
__global__ __launch_bounds__(256) void tt_embed_kernel(
    const float* __restrict__ x, const float* __restrict__ Wi,
    const float* __restrict__ bi, float* __restrict__ h) {
  size_t idx = (size_t)blockIdx.x * 256 + threadIdx.x;
  if (idx >= (size_t)M_TOK * DD) return;
  int d = (int)(idx & (DD - 1));
  size_t bt = idx >> 10;
  int t = (int)(bt % TT);
  const float* xr = x + bt * IN;
  float s = bi[d];
#pragma unroll
  for (int i = 0; i < IN; ++i) s += xr[i] * Wi[i * DD + d];
  int j2 = (d >> 1) << 1;
  float freq = __expf((float)j2 * (-9.210340371976184f / (float)DD)); // -ln(1e4)/D
  float ang = (float)t * freq;
  s += (d & 1) ? __cosf(ang) : __sinf(ang);
  h[idx] = s;
}

// -------------------------------------------------------------------------
// WMMA GEMM: C[M,N] = A[M,K](lda) @ W(^T) + bias (+residual)(+relu)
//   wIsNK==1: W stored [N,K] ; wIsNK==0: W stored [K,N]
// Block: 256 thr = 8 waves. Block tile 128x64, wave tile 32x32 (2x2 WMMA).
// Register-pipelined staging: next K-tile is loaded into float4 regs while
// the WMMAs for the current tile run; regs are cvt'd to f16 and stored to
// LDS at the top of the next iteration. All dims divide tiles exactly.
// LDS rows padded to 40 halves (80 B) so each half-fragment is one b128 load.
// -------------------------------------------------------------------------
__global__ __launch_bounds__(256) void tt_gemm_wmma_kernel(
    const float* __restrict__ A, int lda,
    const float* __restrict__ W,
    const float* __restrict__ bias,
    const float* __restrict__ residual,
    float* __restrict__ C,
    int M, int N, int K, int wIsNK, int doRelu) {
  __shared__ _Float16 As[128][40];   // [m][k]
  __shared__ _Float16 BsT[64][40];   // [n][k]  (k contiguous)
  const int tid  = threadIdx.x;
  const int lane = tid & 31;
  const int wave = tid >> 5;
  const int wm = wave >> 1;          // 0..3 : 32-row slab
  const int wn = wave & 1;           // 0..1 : 32-col slab
  const int bm = blockIdx.y * 128;
  const int bn = blockIdx.x * 64;
  const int m  = lane & 15;
  const int hi = lane >> 4;
  const int a0 = hi ? 8 : 0, a1 = hi ? 24 : 16;  // unused; A frag uses kb (16 consecutive k's)
  const int kb = hi ? 16 : 0;                    // B frag: 16 consecutive k's

  v8f acc[2][2] = {};

  // staging registers for the in-flight K-tile
  float4 ra[4];         // A: 128 rows x 8 float4 = 1024 / 256 thr
  float4 rb[2];         // B (wIsNK): 64 n x 8 float4 = 512 / 256 thr
  float  rbs[8];        // B (K,N layout): 32 k x 64 n = 2048 / 256 thr

  auto loadA = [&](int kk) {
#pragma unroll
    for (int f = 0; f < 4; ++f) {
      int i = tid + f * 256; int r = i >> 3, c4 = i & 7;
      ra[f] = *(const float4*)(A + (size_t)(bm + r) * lda + kk + c4 * 4);
    }
  };
  auto storeA = [&]() {
#pragma unroll
    for (int f = 0; f < 4; ++f) {
      int i = tid + f * 256; int r = i >> 3, c4 = i & 7;
      *(h4*)&As[r][c4 * 4] = cvt_h4(ra[f]);
    }
  };
  auto loadB = [&](int kk) {
#pragma unroll
    for (int f = 0; f < 2; ++f) {
      int i = tid + f * 256; int n = i >> 3, c4 = i & 7;
      rb[f] = *(const float4*)(W + (size_t)(bn + n) * K + kk + c4 * 4);
    }
  };
  auto storeB = [&]() {
#pragma unroll
    for (int f = 0; f < 2; ++f) {
      int i = tid + f * 256; int n = i >> 3, c4 = i & 7;
      *(h4*)&BsT[n][c4 * 4] = cvt_h4(rb[f]);
    }
  };
  auto loadB0 = [&](int kk) {
#pragma unroll
    for (int f = 0; f < 8; ++f) {
      int i = tid + f * 256; int k = i >> 6, n = i & 63;
      rbs[f] = W[(size_t)(kk + k) * N + bn + n];
    }
  };
  auto storeB0 = [&]() {
#pragma unroll
    for (int f = 0; f < 8; ++f) {
      int i = tid + f * 256; int k = i >> 6, n = i & 63;
      BsT[n][k] = (_Float16)rbs[f];
    }
  };

  // prime the pipeline with tile 0
  loadA(0);
  if (wIsNK) loadB(0); else loadB0(0);

  for (int kk = 0; kk < K; kk += 32) {
    // publish the staged tile
    storeA();
    if (wIsNK) storeB(); else storeB0();
    __syncthreads();

    // issue next tile's global loads (latency hidden behind the WMMAs)
    int kn = kk + 32;
    if (kn < K) {
      __builtin_prefetch(A + (size_t)(bm + (tid >> 1)) * lda + kn + 32, 0, 0);
      loadA(kn);
      if (wIsNK) loadB(kn); else loadB0(kn);
    }

    // fragments: each is two aligned 16-B LDS loads
    v16h af[2], bf[2];
#pragma unroll
    for (int mt = 0; mt < 2; ++mt) {
      const _Float16* ap = &As[wm * 32 + mt * 16 + m][0];
      af[mt] = cat16(*(const h8*)(ap + kb), *(const h8*)(ap + kb + 8));
    }
#pragma unroll
    for (int nt = 0; nt < 2; ++nt) {
      const _Float16* bp = &BsT[wn * 32 + nt * 16 + m][kb];
      bf[nt] = cat16(*(const h8*)(bp), *(const h8*)(bp + 8));
    }
#pragma unroll
    for (int mt = 0; mt < 2; ++mt)
#pragma unroll
      for (int nt = 0; nt < 2; ++nt)
        acc[mt][nt] = __builtin_amdgcn_wmma_f32_16x16x32_f16(
            false, af[mt], false, bf[nt], (short)0, acc[mt][nt], false, false);
    __syncthreads();   // all waves done reading LDS before next store
  }

  // ---- epilogue
#pragma unroll
  for (int mt = 0; mt < 2; ++mt) {
#pragma unroll
    for (int nt = 0; nt < 2; ++nt) {
      int col = bn + wn * 32 + nt * 16 + m;
      float bv = bias[col];
#pragma unroll
      for (int r = 0; r < 8; ++r) {
        int row = bm + wm * 32 + mt * 16 + r + (hi << 3);
        float fv = acc[mt][nt][r] + bv;
        if (residual) fv += residual[(size_t)row * N + col];
        if (doRelu) fv = fmaxf(fv, 0.0f);
        C[(size_t)row * N + col] = fv;
      }
    }
  }
}

// -------------------------------------------------------------------------
// Fused attention, one block per (b, head). qkv layout: [B*T][3*D].
// (Reference mask adds a per-query constant to every key -> softmax no-op.)
// Context written back into the Q columns of qkv (dead after score phase).
// Dynamic LDS: Ks[112][136] f16, VsT[128][136] f16 (transposed), Sc[112][136] f32.
// -------------------------------------------------------------------------
#define SST 136
__global__ __launch_bounds__(256) void tt_attn_kernel(float* __restrict__ qkv) {
  extern __shared__ char smem[];
  _Float16* Ks  = (_Float16*)smem;             // [key][hd]
  _Float16* VsT = Ks + 112 * SST;              // [hd][key]
  float*    Sc  = (float*)(VsT + 128 * SST);   // [q][key]

  const int bI   = blockIdx.x >> 3;
  const int hh   = blockIdx.x & 7;
  const int tid  = threadIdx.x;
  const int lane = tid & 31;
  const int wave = tid >> 5;
  const int m    = lane & 15;
  const int hi   = lane >> 4;
  const int a0f  = hi ? 8 : 0, a1f = hi ? 24 : 16;
  const int kbf  = hi ? 16 : 0;
  const size_t rowBase = (size_t)bI * TT * (3 * DD);

  // stage K [112 keys][128 hd] (zero-padded past T)
  for (int i = tid; i < 112 * 32; i += 256) {
    int t = i >> 5, c4 = i & 31;
    float4 f = make_float4(0.f, 0.f, 0.f, 0.f);
    if (t < TT)
      f = *(const float4*)(qkv + rowBase + (size_t)t * (3 * DD) + DD + hh * HD + c4 * 4);
    *(h4*)&Ks[t * SST + c4 * 4] = cvt_h4(f);
  }
  // stage V transposed: VsT[hd][key], keys padded to 128
  for (int i = tid; i < 128 * 32; i += 256) {
    int t = i >> 5, c4 = i & 31;
    float4 f = make_float4(0.f, 0.f, 0.f, 0.f);
    if (t < TT)
      f = *(const float4*)(qkv + rowBase + (size_t)t * (3 * DD) + 2 * DD + hh * HD + c4 * 4);
    VsT[(c4 * 4 + 0) * SST + t] = (_Float16)f.x;
    VsT[(c4 * 4 + 1) * SST + t] = (_Float16)f.y;
    VsT[(c4 * 4 + 2) * SST + t] = (_Float16)f.z;
    VsT[(c4 * 4 + 3) * SST + t] = (_Float16)f.w;
  }
  __syncthreads();

  // ---- scores: waves 0..6 own 16 query rows each
  if (wave < 7) {
    v16h aq[4];
    int qrow = wave * 16 + m;
    int qsrc = (qrow < TT) ? qrow : 0;   // pad rows read row 0 (discarded later)
    const float* qp = qkv + rowBase + (size_t)qsrc * (3 * DD) + hh * HD;
#pragma unroll
    for (int ks = 0; ks < 4; ++ks)
      aq[ks] = pack16(*(const float4*)(qp + ks * 32 + kbf),
                      *(const float4*)(qp + ks * 32 + kbf + 4),
                      *(const float4*)(qp + ks * 32 + kbf + 8),
                      *(const float4*)(qp + ks * 32 + kbf + 12));
    for (int kt = 0; kt < 7; ++kt) {     // key tiles 0..111
      v8f acc = {};
      int key = kt * 16 + m;
#pragma unroll
      for (int ks = 0; ks < 4; ++ks) {   // HD = 4 * 32
        const _Float16* kp = Ks + key * SST + ks * 32 + kbf;
        v16h bfr = cat16(*(const h8*)(kp), *(const h8*)(kp + 8));
        acc = __builtin_amdgcn_wmma_f32_16x16x32_f16(
            false, aq[ks], false, bfr, (short)0, acc, false, false);
      }
#pragma unroll
      for (int r = 0; r < 8; ++r) {
        int row = wave * 16 + r + (hi << 3);
        Sc[row * SST + kt * 16 + m] = acc[r] * SCALE;
      }
    }
  }
  __syncthreads();

  // ---- softmax over the 100 valid keys; zero padded keys 100..127
  if (tid < 112) {
    float* sr = Sc + tid * SST;
    float mx = -1e30f;
    for (int c = 0; c < TT; ++c) mx = fmaxf(mx, sr[c]);
    float sum = 0.0f;
    for (int c = 0; c < TT; ++c) { float e = __expf(sr[c] - mx); sr[c] = e; sum += e; }
    float inv = 1.0f / sum;
    for (int c = 0; c < TT; ++c) sr[c] *= inv;
    for (int c = TT; c < 128; ++c) sr[c] = 0.0f;
  }
  __syncthreads();

  // ---- context = attn @ V ; write into Q columns of qkv
  if (wave < 7) {
    v16h aa[4];
    int qrow = wave * 16 + m;
    const float* sp = Sc + qrow * SST;
#pragma unroll
    for (int ks = 0; ks < 4; ++ks)
      aa[ks] = pack16(*(const float4*)(sp + ks * 32 + kbf),
                      *(const float4*)(sp + ks * 32 + kbf + 4),
                      *(const float4*)(sp + ks * 32 + kbf + 8),
                      *(const float4*)(sp + ks * 32 + kbf + 12));
#pragma unroll
    for (int nt = 0; nt < 8; ++nt) {     // HD = 8 * 16 output cols
      v8f acc = {};
      int hd = nt * 16 + m;
#pragma unroll
      for (int ks = 0; ks < 4; ++ks) {   // keys 0..127 (padded attn = 0)
        const _Float16* vp = VsT + hd * SST + ks * 32 + kbf;
        v16h bfr = cat16(*(const h8*)(vp), *(const h8*)(vp + 8));
        acc = __builtin_amdgcn_wmma_f32_16x16x32_f16(
            false, aa[ks], false, bfr, (short)0, acc, false, false);
      }
#pragma unroll
      for (int r = 0; r < 8; ++r) {
        int row = wave * 16 + r + (hi << 3);
        if (row < TT)
          qkv[rowBase + (size_t)row * (3 * DD) + hh * HD + nt * 16 + m] = acc[r];
      }
    }
  }
}

// -------------------------------------------------------------------------
// LayerNorm (in place), one block per token row of 1024. float4 I/O.
// -------------------------------------------------------------------------
__global__ __launch_bounds__(256) void tt_ln_kernel(
    float* __restrict__ h, const float* __restrict__ g,
    const float* __restrict__ bta) {
  __shared__ float red[256];
  float4* p = (float4*)(h + (size_t)blockIdx.x * DD);
  float4 v = p[threadIdx.x];
  float s = v.x + v.y + v.z + v.w;
  red[threadIdx.x] = s;
  __syncthreads();
  for (int off = 128; off; off >>= 1) {
    if (threadIdx.x < off) red[threadIdx.x] += red[threadIdx.x + off];
    __syncthreads();
  }
  float mean = red[0] * (1.0f / (float)DD);
  __syncthreads();
  float dx = v.x - mean, dy = v.y - mean, dz = v.z - mean, dw = v.w - mean;
  red[threadIdx.x] = dx * dx + dy * dy + dz * dz + dw * dw;
  __syncthreads();
  for (int off = 128; off; off >>= 1) {
    if (threadIdx.x < off) red[threadIdx.x] += red[threadIdx.x + off];
    __syncthreads();
  }
  float inv = rsqrtf(red[0] * (1.0f / (float)DD) + 1e-5f);
  __syncthreads();
  float4 gg = ((const float4*)g)[threadIdx.x];
  float4 bb = ((const float4*)bta)[threadIdx.x];
  float4 o;
  o.x = dx * inv * gg.x + bb.x;
  o.y = dy * inv * gg.y + bb.y;
  o.z = dz * inv * gg.z + bb.z;
  o.w = dw * inv * gg.w + bb.w;
  p[threadIdx.x] = o;
}

// -------------------------------------------------------------------------
extern "C" void kernel_launch(void* const* d_in, const int* in_sizes, int n_in,
                              void* d_out, int out_size, void* d_ws, size_t ws_size,
                              hipStream_t stream) {
  const float* x     = (const float*)d_in[0];
  const float* Wi    = (const float*)d_in[1];
  const float* bi    = (const float*)d_in[2];
  const float* qkv_w = (const float*)d_in[3];
  const float* qkv_b = (const float*)d_in[4];
  const float* out_w = (const float*)d_in[5];
  const float* out_b = (const float*)d_in[6];
  const float* ff1_w = (const float*)d_in[7];
  const float* ff1_b = (const float*)d_in[8];
  const float* ff2_w = (const float*)d_in[9];
  const float* ff2_b = (const float*)d_in[10];
  const float* ln1_g = (const float*)d_in[11];
  const float* ln1_b = (const float*)d_in[12];
  const float* ln2_g = (const float*)d_in[13];
  const float* ln2_b = (const float*)d_in[14];
  const float* Wo    = (const float*)d_in[15];
  const float* bo    = (const float*)d_in[16];
  float* out = (float*)d_out;

  float* h   = (float*)d_ws;                  // [25600][1024]
  float* qkv = h + (size_t)M_TOK * DD;        // [25600][3072] (also FFN scratch)

  {
    size_t tot = (size_t)M_TOK * DD;
    tt_embed_kernel<<<dim3((unsigned)((tot + 255) / 256)), 256, 0, stream>>>(x, Wi, bi, h);
  }

  const size_t attn_lds = (size_t)(112 * SST + 128 * SST) * 2 + (size_t)112 * SST * 4;

  for (int l = 0; l < LL; ++l) {
    // QKV projection
    tt_gemm_wmma_kernel<<<dim3(3 * DD / 64, M_TOK / 128), 256, 0, stream>>>(
        h, DD, qkv_w + (size_t)l * 3 * DD * DD, qkv_b + (size_t)l * 3 * DD,
        nullptr, qkv, M_TOK, 3 * DD, DD, 1, 0);
    // fused attention (ctx -> Q columns of qkv)
    tt_attn_kernel<<<dim3(BB * HH), 256, attn_lds, stream>>>(qkv);
    // output projection + residual -> h
    tt_gemm_wmma_kernel<<<dim3(DD / 64, M_TOK / 128), 256, 0, stream>>>(
        qkv, 3 * DD, out_w + (size_t)l * DD * DD, out_b + (size_t)l * DD,
        h, h, M_TOK, DD, DD, 1, 0);
    tt_ln_kernel<<<dim3(M_TOK), 256, 0, stream>>>(h, ln1_g + (size_t)l * DD, ln1_b + (size_t)l * DD);
    // FFN
    tt_gemm_wmma_kernel<<<dim3(DD / 64, M_TOK / 128), 256, 0, stream>>>(
        h, DD, ff1_w + (size_t)l * DD * DD, ff1_b + (size_t)l * DD,
        nullptr, qkv, M_TOK, DD, DD, 1, 1);
    tt_gemm_wmma_kernel<<<dim3(DD / 64, M_TOK / 128), 256, 0, stream>>>(
        qkv, DD, ff2_w + (size_t)l * DD * DD, ff2_b + (size_t)l * DD,
        h, h, M_TOK, DD, DD, 1, 0);
    tt_ln_kernel<<<dim3(M_TOK), 256, 0, stream>>>(h, ln2_g + (size_t)l * DD, ln2_b + (size_t)l * DD);
  }

  // head: [256, 102400] @ Wo[102400, 256] + bo
  tt_gemm_wmma_kernel<<<dim3(OUT / 64, BB / 128), 256, 0, stream>>>(
      h, TT * DD, Wo, bo, nullptr, out, BB, OUT, TT * DD, 0, 0);
}
